// GCAE_25048249270387
// MI455X (gfx1250) — matrix operand order repaired and checked by reading, not compiled, not run on hardware
//
#include <hip/hip_runtime.h>

// ---------------------------------------------------------------------------
// GCN autoencoder for MI455X (gfx1250, wave32).
//  - GEMMs: v_wmma_f32_16x16x32_bf16, hi/lo bf16 split (3 WMMAs / tile pair)
//    for ~fp32 accuracy, f32 accumulate.
//  - A and W are pre-packed into WMMA fragment layout (bf16 hi/lo) so the
//    GEMM inner loop is pure b128 loads + WMMA (no conversion VALU, no
//    strided scalar gathers).
//  - Aggregation: precomputed per-edge norm + hardware f32 scatter atomics.
// ---------------------------------------------------------------------------

typedef __attribute__((ext_vector_type(16))) __bf16 v16bf;
typedef __attribute__((ext_vector_type(8)))  float  v8f;

// -------------------- degree / normalization --------------------
__global__ void k_deg_init(float* __restrict__ deg, int n) {
  int i = blockIdx.x * blockDim.x + threadIdx.x;
  if (i < n) deg[i] = 1.0f;                       // self-loop contribution
}

__global__ void k_deg_count(const int* __restrict__ dst, float* __restrict__ deg, int e) {
  int i = blockIdx.x * blockDim.x + threadIdx.x;
  if (i < e) unsafeAtomicAdd(&deg[dst[i]], 1.0f); // global_atomic_add_f32
}

__global__ void k_dinv(float* __restrict__ d, int n) {
  int i = blockIdx.x * blockDim.x + threadIdx.x;
  if (i < n) d[i] = rsqrtf(d[i]);                 // deg >= 1 always
}

__global__ void k_edge_norm(const int* __restrict__ src, const int* __restrict__ dst,
                            const float* __restrict__ dinv, float* __restrict__ norm, int e) {
  int i = blockIdx.x * blockDim.x + threadIdx.x;
  if (i < e) norm[i] = dinv[src[i]] * dinv[dst[i]];
}

// out[i,c] = h[i,c]*dinv[i]^2 + b[c]   (self-loop term + bias; inits out)
__global__ void k_gcn_self(const float* __restrict__ h, const float* __restrict__ dinv,
                           const float* __restrict__ b, float* __restrict__ out,
                           int n, int C) {
  int i = blockIdx.x * blockDim.x + threadIdx.x;
  if (i < n * C) {
    int r = i / C, c = i - r * C;
    float di = dinv[r];
    out[i] = h[i] * di * di + b[c];
  }
}

// out[dst,c] += h[src,c] * norm[e]   (scatter-add over edges)
__global__ void k_gcn_scatter(const float* __restrict__ h, const int* __restrict__ src,
                              const int* __restrict__ dst, const float* __restrict__ norm,
                              float* __restrict__ out, int total, int C) {
  int i = blockIdx.x * blockDim.x + threadIdx.x;
  if (i < total) {
    int e = i / C, c = i - e * C;
    int s = src[e], d = dst[e];
    unsafeAtomicAdd(&out[(size_t)d * C + c], h[(size_t)s * C + c] * norm[e]);
  }
}

__global__ void k_relu(float* __restrict__ x, int n) {
  int i = blockIdx.x * blockDim.x + threadIdx.x;
  if (i < n) x[i] = fmaxf(x[i], 0.0f);
}

// -------------------- fragment packing (f32 -> bf16 hi/lo) -----------------
// A-fragment layout (16-bit A 16x32):
//   entry = ((mt*ksteps + kt)*32 + lane)*16 + e
//   m = mt*16 + (lane&15);  k = kt*32 + 8*(lane>>4) + (e<8 ? e : e+8)
__global__ void k_pack_a(const float* __restrict__ X, __bf16* __restrict__ hi,
                         __bf16* __restrict__ lo, int M, int K) {
  int i = blockIdx.x * blockDim.x + threadIdx.x;
  if (i >= M * K) return;
  int e    = i & 15;
  int lane = (i >> 4) & 31;
  int t    = i >> 9;                 // mt*ksteps + kt
  int ksteps = K >> 5;
  int mt = t / ksteps, kt = t - mt * ksteps;
  int m = mt * 16 + (lane & 15);
  int k = kt * 32 + ((lane >> 4) << 3) + (e < 8 ? e : e + 8);
  float v = X[(size_t)m * K + k];
  __bf16 h = (__bf16)v;
  hi[i] = h;
  lo[i] = (__bf16)(v - (float)h);
}

// B-fragment layout (16-bit B 32x16):
//   entry = ((kt*ntiles + nt)*32 + lane)*16 + e
//   n = nt*16 + (lane&15);  k = kt*32 + 8*(lane>>4) + (e<8 ? e : e+8)
__global__ void k_pack_w(const float* __restrict__ W, __bf16* __restrict__ hi,
                         __bf16* __restrict__ lo, int K, int N) {
  int i = blockIdx.x * blockDim.x + threadIdx.x;
  if (i >= K * N) return;
  int e    = i & 15;
  int lane = (i >> 4) & 31;
  int t    = i >> 9;                 // kt*ntiles + nt
  int ntiles = N >> 4;
  int kt = t / ntiles, nt = t - kt * ntiles;
  int n = nt * 16 + (lane & 15);
  int k = kt * 32 + ((lane >> 4) << 3) + (e < 8 ? e : e + 8);
  float v = W[(size_t)k * N + n];
  __bf16 h = (__bf16)v;
  hi[i] = h;
  lo[i] = (__bf16)(v - (float)h);
}

// -------------------- WMMA GEMM: Y[M,N] = X[M,K] @ W[K,N] (+bias) ----------
// Inputs pre-packed in fragment layout. Block = 128 threads (4 waves);
// each wave computes a 16(M) x 64(N) strip. K%32==0, N%64==0, M%16==0.
__global__ __launch_bounds__(128)
void k_gemm_wmma(const __bf16* __restrict__ Ahi_, const __bf16* __restrict__ Alo_,
                 const __bf16* __restrict__ Whi_, const __bf16* __restrict__ Wlo_,
                 const float* __restrict__ bias, float* __restrict__ Y,
                 int M, int K, int N) {
  const int lane = threadIdx.x & 31;
  const int wave = threadIdx.x >> 5;
  const int mt   = blockIdx.x * 4 + wave;
  const int m0   = mt * 16;
  const int n0   = blockIdx.y * 64;
  if (m0 >= M) return;               // uniform per wave (EXEC stays all-1s)

  const int ksteps = K >> 5;
  const int ntiles = N >> 4;
  const v16bf* __restrict__ Ah = (const v16bf*)Ahi_;
  const v16bf* __restrict__ Al = (const v16bf*)Alo_;
  const v16bf* __restrict__ Bh = (const v16bf*)Whi_;
  const v16bf* __restrict__ Bl = (const v16bf*)Wlo_;

  v8f acc[4] = {};

  size_t ta = (size_t)mt * ksteps * 32 + lane;
  for (int kt = 0; kt < ksteps; ++kt, ta += 32) {
    v16bf ah = Ah[ta];
    v16bf al = Al[ta];
    size_t tb = ((size_t)kt * ntiles + (n0 >> 4)) * 32 + lane;
#pragma unroll
    for (int j = 0; j < 4; ++j) {
      v16bf bh = Bh[tb + (size_t)j * 32];
      v16bf bl = Bl[tb + (size_t)j * 32];
      // fp32-emulating 3-term split: Ah*Bh + Al*Bh + Ah*Bl
      acc[j] = __builtin_amdgcn_wmma_f32_16x16x32_bf16(false, ah, false, bh,
                                                       (short)0, acc[j], false, false);
      acc[j] = __builtin_amdgcn_wmma_f32_16x16x32_bf16(false, al, false, bh,
                                                       (short)0, acc[j], false, false);
      acc[j] = __builtin_amdgcn_wmma_f32_16x16x32_bf16(false, ah, false, bl,
                                                       (short)0, acc[j], false, false);
    }
  }

  // D layout: VGPR r -> (M = m0 + r + 8*half, N = n0 + 16j + lm)
  const int half  = lane >> 4;
  const int lm    = lane & 15;
  const int rbase = m0 + 8 * half;
#pragma unroll
  for (int j = 0; j < 4; ++j) {
    int col = n0 + 16 * j + lm;
    float bv = bias ? bias[col] : 0.0f;
#pragma unroll
    for (int r = 0; r < 8; ++r)
      Y[(size_t)(rbase + r) * N + col] = acc[j][r] + bv;
  }
}

// ---------------------------------------------------------------------------
extern "C" void kernel_launch(void* const* d_in, const int* in_sizes, int n_in,
                              void* d_out, int out_size, void* d_ws, size_t ws_size,
                              hipStream_t stream) {
  (void)n_in; (void)out_size; (void)ws_size;

  const float* x   = (const float*)d_in[0];
  const int*   ei  = (const int*)  d_in[1];
  const float* eW1 = (const float*)d_in[2];
  const float* eb1 = (const float*)d_in[3];
  const float* eW2 = (const float*)d_in[4];
  const float* eb2 = (const float*)d_in[5];
  const float* eWf = (const float*)d_in[6];
  const float* ebf = (const float*)d_in[7];
  const float* dW1 = (const float*)d_in[8];
  const float* db1 = (const float*)d_in[9];
  const float* dW2 = (const float*)d_in[10];
  const float* db2 = (const float*)d_in[11];
  const float* dWf = (const float*)d_in[12];
  const float* dbf = (const float*)d_in[13];

  const int Nn = in_sizes[0] / 128;  // 50000 nodes
  const int E  = in_sizes[1] / 2;    // 800000 edges
  const int* src = ei;
  const int* dst = ei + E;

  // workspace layout (all 32B-aligned):
  //   dinv(N) | norm(E) | bufA(N*256 f32) | bufB(N*256 f32)
  //   | Apk_hi(N*256 bf16) | Apk_lo | Wpk_hi(128K bf16) | Wpk_lo     ~158 MB
  float* dinv = (float*)d_ws;
  float* norm = dinv + (((size_t)Nn + 255) & ~(size_t)255);
  float* bufA = norm + (((size_t)E + 255) & ~(size_t)255);
  float* bufB = bufA + (size_t)Nn * 256;
  __bf16* Apk_hi = (__bf16*)(bufB + (size_t)Nn * 256);
  __bf16* Apk_lo = Apk_hi + (size_t)Nn * 256;
  __bf16* Wpk_hi = Apk_lo + (size_t)Nn * 256;
  __bf16* Wpk_lo = Wpk_hi + 131072;         // max K*N = 128*1024

  auto cdiv = [](int a, int b) { return (a + b - 1) / b; };

  // --- degree / normalization (shared by all 4 GCN layers) ---
  k_deg_init <<<cdiv(Nn, 256), 256, 0, stream>>>(dinv, Nn);
  k_deg_count<<<cdiv(E, 256), 256, 0, stream>>>(dst, dinv, E);
  k_dinv     <<<cdiv(Nn, 256), 256, 0, stream>>>(dinv, Nn);
  k_edge_norm<<<cdiv(E, 256), 256, 0, stream>>>(src, dst, dinv, norm, E);

  auto gemm = [&](const float* Xp, const float* Wp, const float* bp, float* Yp,
                  int K, int N) {
    k_pack_a<<<cdiv(Nn * K, 256), 256, 0, stream>>>(Xp, Apk_hi, Apk_lo, Nn, K);
    k_pack_w<<<cdiv(K * N, 256), 256, 0, stream>>>(Wp, Wpk_hi, Wpk_lo, K, N);
    dim3 grid(cdiv(cdiv(Nn, 16), 4), N / 64);
    k_gemm_wmma<<<grid, 128, 0, stream>>>(Apk_hi, Apk_lo, Wpk_hi, Wpk_lo,
                                          bp, Yp, Nn, K, N);
  };
  auto gcn_agg = [&](const float* hp, const float* bp, float* op, int C, bool relu) {
    k_gcn_self   <<<cdiv(Nn * C, 256), 256, 0, stream>>>(hp, dinv, bp, op, Nn, C);
    k_gcn_scatter<<<cdiv(E * C, 256), 256, 0, stream>>>(hp, src, dst, norm, op, E * C, C);
    if (relu) k_relu<<<cdiv(Nn * C, 256), 256, 0, stream>>>(op, Nn * C);
  };

  // --- encoder ---
  gemm(x,    eW1, nullptr, bufA, 128, 128);    // h = x @ eW1
  gcn_agg(bufA, eb1, bufB, 128, true);         // relu(gcn aggregate)
  gemm(bufB, eW2, nullptr, bufA, 128, 64);
  gcn_agg(bufA, eb2, bufB, 64, false);
  gemm(bufB, eWf, ebf, bufA, 64, 64);          // z = h @ eWf + ebf

  // --- decoder ---
  gemm(bufA, dW1, nullptr, bufB, 64, 256);
  gcn_agg(bufB, db1, bufA, 256, true);
  gemm(bufA, dW2, nullptr, bufB, 256, 128);
  gcn_agg(bufB, db2, bufA, 128, false);
  gemm(bufA, dWf, dbf, (float*)d_out, 128, 1024);  // x_hat [N,1024]
}
